// NMNISTReLUNetwork_25898652795331
// MI455X (gfx1250) — matrix operand order repaired and checked by reading, not compiled, not run on hardware
//
#include <hip/hip_runtime.h>
#include <hip/hip_bf16.h>

// ---------------------------------------------------------------------------
// MI455X (gfx1250) implementation: implicit-GEMM conv3d on the WMMA pipe.
// bf16 activations/weights, f32 accumulation via v_wmma_f32_16x16x32_bf16.
// Each wave register-blocks a 16(m) x 64(n) tile: 4 accumulators share one
// A (weight) fragment per K-step to raise arithmetic intensity.
// ---------------------------------------------------------------------------

typedef __attribute__((ext_vector_type(16))) __bf16 v16bf;
typedef __attribute__((ext_vector_type(8)))  __bf16 bf16x8;
typedef __attribute__((ext_vector_type(8)))  float  v8f;

namespace nm {
constexpr int B  = 16;
constexpr int T  = 128;
constexpr int Cx = 2;
constexpr int H  = 34;
constexpr int W  = 34;
constexpr int HW = H * W;              // 1156
constexpr int POS1 = B * T * H * W;    // 2,367,488  (conv1 output positions)
constexpr int T2 = 64, H2 = 17, W2 = 17;
constexpr int SPT2 = T2 * H2 * W2;     // 18,496
constexpr int POS2 = B * SPT2;         // 295,936    (conv2/3 output positions)
}

__device__ __forceinline__ bf16x8 zero8() {
  bf16x8 z;
#pragma unroll
  for (int i = 0; i < 8; ++i) z[i] = (__bf16)0.0f;
  return z;
}

// ---------------------------------------------------------------------------
// prep: x (B,T,2,H,W) f32 -> xp (B,T,H,W,2) bf16 ; mask1 (B,T,H,W) u8
// ---------------------------------------------------------------------------
__global__ __launch_bounds__(256) void prep_x_mask(
    const float* __restrict__ x, __bf16* __restrict__ xp,
    unsigned char* __restrict__ m1) {
  using namespace nm;
  unsigned idx = blockIdx.x * 256u + threadIdx.x;   // exact grid
  unsigned b = idx / (unsigned)(T * HW);
  unsigned r = idx % (unsigned)(T * HW);
  unsigned t = r / (unsigned)HW;
  unsigned hw = r % (unsigned)HW;
  unsigned base = ((b * T + t) * Cx) * HW + hw;     // c stride = HW
  float x0 = x[base];
  float x1 = x[base + HW];
  xp[(size_t)idx * 2 + 0] = (__bf16)x0;
  xp[(size_t)idx * 2 + 1] = (__bf16)x1;
  m1[idx] = (fabsf(x0) + fabsf(x1)) > 0.0f ? 1 : 0;
}

// ---------------------------------------------------------------------------
// 2x2x2 max-pool of mask1 -> mask2 (B,64,17,17) u8
// ---------------------------------------------------------------------------
__global__ __launch_bounds__(256) void prep_pool_mask(
    const unsigned char* __restrict__ m1, unsigned char* __restrict__ m2) {
  using namespace nm;
  unsigned idx = blockIdx.x * 256u + threadIdx.x;   // exact grid (POS2)
  unsigned b  = idx / (unsigned)SPT2;
  unsigned r  = idx % (unsigned)SPT2;
  unsigned tp = r / (unsigned)(H2 * W2);
  unsigned r2 = r % (unsigned)(H2 * W2);
  unsigned hp = r2 / (unsigned)W2;
  unsigned wp = r2 % (unsigned)W2;
  unsigned char v = 0;
#pragma unroll
  for (int dt = 0; dt < 2; ++dt)
#pragma unroll
    for (int dh = 0; dh < 2; ++dh)
#pragma unroll
      for (int dw = 0; dw < 2; ++dw)
        v |= m1[((b * T + (2 * tp + dt)) * H + (2 * hp + dh)) * W + (2 * wp + dw)];
  m2[idx] = v ? 1 : 0;
}

// ---------------------------------------------------------------------------
// per-batch mask2 count (cnt = max(sum,1)) + zero the pooled accumulator
// ---------------------------------------------------------------------------
__global__ __launch_bounds__(256) void prep_count_zero(
    const unsigned char* __restrict__ m2, float* __restrict__ cnt,
    float* __restrict__ pooled) {
  using namespace nm;
  __shared__ int s;
  int b = blockIdx.x;
  int tid = threadIdx.x;
  if (tid == 0) s = 0;
  __syncthreads();
  int c = 0;
  for (int i = tid; i < SPT2; i += 256) c += m2[b * SPT2 + i];
  atomicAdd(&s, c);
  __syncthreads();
  if (tid == 0) cnt[b] = fmaxf((float)s, 1.0f);
  if (tid < 128) pooled[b * 128 + tid] = 0.0f;
}

// ---------------------------------------------------------------------------
// weight repack: OIDHW f32 -> bf16 [cout][KPAD], k = tap*CIN + c, zero-padded
// ---------------------------------------------------------------------------
__global__ __launch_bounds__(256) void prep_w(
    const float* __restrict__ src, __bf16* __restrict__ dst,
    int CIN, int COUT, int TAPS, int KPAD) {
  int idx = blockIdx.x * 256 + threadIdx.x;
  if (idx >= COUT * KPAD) return;
  int m = idx / KPAD;
  int k = idx % KPAD;
  if (k < TAPS * CIN) {
    int tap = k / CIN;
    int c   = k % CIN;
    dst[idx] = (__bf16)src[(m * CIN + c) * TAPS + tap];
  } else {
    dst[idx] = (__bf16)0.0f;
  }
}

// ---------------------------------------------------------------------------
// Implicit-GEMM conv3d with WMMA. One wave = one 16(m) x 64(n) output tile
// (4 n-subtiles sharing each A fragment).
//   A (weights):  [COUT][KPAD] bf16 rows; A-fragment k = 16*lh + s (contiguous order) = 2 contiguous loads
//   B (patches):  NDHWC bf16; for CIN%16==0 a lane's 16 K-values are one tap
//                 -> 2 contiguous 16B loads (zero-filled outside the tensor)
// POOL=true fuses relu*mask + global pooling via LDS + global f32 atomics.
// ---------------------------------------------------------------------------
template <int CIN, int COUT, int KD, int KH, int KW, int S, int P,
          int Ti, int Hi, int Wi, int To, int Ho, int Wo,
          int KTRUE, int KPAD, int MT, bool RELU, bool POOL>
__global__ __launch_bounds__(256) void conv_wmma(
    const __bf16* __restrict__ in, const __bf16* __restrict__ wp,
    const unsigned char* __restrict__ mk, __bf16* __restrict__ out,
    float* __restrict__ pooled) {
  constexpr int NPW = 8 / MT;                 // n-groups per 8-wave block
  constexpr int NB  = 4;                      // n-subtiles per wave (64 pos)
  constexpr int SPT = To * Ho * Wo;           // divisible by 64 for all layers
  constexpr bool KVEC = (CIN % 16) == 0;

  __shared__ float poolAcc[POOL ? COUT : 1];

  const unsigned tid  = threadIdx.x;
  const unsigned wv   = tid >> 5;
  const unsigned lane = tid & 31;
  const unsigned laneN = lane & 15;
  const unsigned lh    = lane >> 4;           // lane half (0/1)

  if (POOL) {
    for (unsigned i = tid; i < (unsigned)COUT; i += 256) poolAcc[i] = 0.0f;
    __syncthreads();
  }

  const unsigned mBase = (wv % MT) * 16;
  const unsigned nGrp  = blockIdx.x * NPW + (wv / MT);
  const unsigned nBase = nGrp * 64;           // wave's first position
  const unsigned b     = nBase / (unsigned)SPT;   // wave-uniform batch index

  unsigned n_[NB];
  int to_[NB], ho_[NB], wo_[NB];
#pragma unroll
  for (int j = 0; j < NB; ++j) {
    const unsigned n = nBase + j * 16 + laneN;
    n_[j] = n;
    unsigned r = n - b * (unsigned)SPT;       // same b across the group
    to_[j] = (int)(r / (unsigned)(Ho * Wo));  r %= (unsigned)(Ho * Wo);
    ho_[j] = (int)(r / (unsigned)Wo);
    wo_[j] = (int)(r % (unsigned)Wo);
  }

  const __bf16* wRow = wp + (size_t)(mBase + laneN) * KPAD;

  v8f acc[NB];
#pragma unroll
  for (int j = 0; j < NB; ++j)
#pragma unroll
    for (int i = 0; i < 8; ++i) acc[j][i] = 0.0f;

  if (KVEC) {
    // --- nested tap loops: no div/mod, hoisted bounds checks -------------
    int k0 = 0;
#pragma unroll 1
    for (int kd = 0; kd < KD; ++kd) {
      int ti_[NB];
      bool tok_[NB];
#pragma unroll
      for (int j = 0; j < NB; ++j) {
        ti_[j] = to_[j] * S + kd - P;
        tok_[j] = (unsigned)ti_[j] < (unsigned)Ti;
      }
#pragma unroll 1
      for (int kh = 0; kh < KH; ++kh) {
        int hi_[NB];
        bool thok_[NB];
#pragma unroll
        for (int j = 0; j < NB; ++j) {
          hi_[j] = ho_[j] * S + kh - P;
          thok_[j] = tok_[j] && ((unsigned)hi_[j] < (unsigned)Hi);
        }
#pragma unroll
        for (int kw = 0; kw < KW; ++kw) {
#pragma unroll
          for (int c0 = 0; c0 < CIN; c0 += 32) {
            const __bf16* ap = wRow + (k0 + lh * 16);
            bf16x8 a_lo = *(const bf16x8*)(ap);
            bf16x8 a_hi = *(const bf16x8*)(ap + 8);
            v16bf a = __builtin_shufflevector(a_lo, a_hi,
                0,1,2,3,4,5,6,7,8,9,10,11,12,13,14,15);

#pragma unroll
            for (int j = 0; j < NB; ++j) {
              const int wi = wo_[j] * S + kw - P;
              const bool ok = thok_[j] && ((unsigned)wi < (unsigned)Wi);
              bf16x8 blo = zero8(), bhi = zero8();
              if (ok) {
                const size_t base =
                    ((((size_t)b * Ti + ti_[j]) * Hi + hi_[j]) * Wi + wi) *
                        CIN + (c0 + lh * 16);
                const bf16x8* bp = (const bf16x8*)(in + base);
                blo = bp[0];
                bhi = bp[1];
              }
              v16bf bf = __builtin_shufflevector(blo, bhi,
                  0,1,2,3,4,5,6,7,8,9,10,11,12,13,14,15);
              acc[j] = __builtin_amdgcn_wmma_f32_16x16x32_bf16(
                  false, a, false, bf, (short)0, acc[j], false, false);
            }
            k0 += 32;
          }
        }
      }
    }
  } else {
    // --- small-CIN (conv1) scalar-gather path ----------------------------
#pragma unroll 1
    for (int k0 = 0; k0 < KPAD; k0 += 32) {
      const __bf16* ap = wRow + k0 + lh * 16;
      bf16x8 a_lo = *(const bf16x8*)(ap);
      bf16x8 a_hi = *(const bf16x8*)(ap + 8);
      v16bf a = __builtin_shufflevector(a_lo, a_hi,
          0,1,2,3,4,5,6,7,8,9,10,11,12,13,14,15);

#pragma unroll
      for (int j = 0; j < NB; ++j) {
        v16bf bf;
#pragma unroll
        for (int e = 0; e < 16; ++e) {
          const unsigned k = (unsigned)k0 + lh * 16 + e;
          __bf16 v = (__bf16)0.0f;
          if (k < (unsigned)KTRUE) {
            const unsigned tap = k / (unsigned)CIN;
            const unsigned c   = k % (unsigned)CIN;
            const int kd  = (int)(tap / (unsigned)(KH * KW));
            const unsigned rem = tap % (unsigned)(KH * KW);
            const int kh  = (int)(rem / (unsigned)KW);
            const int kw  = (int)(rem % (unsigned)KW);
            const int ti = to_[j] * S + kd - P;
            const int hi = ho_[j] * S + kh - P;
            const int wi = wo_[j] * S + kw - P;
            if ((unsigned)ti < (unsigned)Ti && (unsigned)hi < (unsigned)Hi &&
                (unsigned)wi < (unsigned)Wi)
              v = in[((((size_t)b * Ti + ti) * Hi + hi) * Wi + wi) * CIN + c];
          }
          bf[e] = v;
        }
        acc[j] = __builtin_amdgcn_wmma_f32_16x16x32_bf16(
            false, a, false, bf, (short)0, acc[j], false, false);
      }
    }
  }

  // ---- epilogue: mask (+relu), store bf16 NDHWC or fused pooling ----
  if (POOL) {
#pragma unroll
    for (int j = 0; j < NB; ++j) {
      const float mval = mk[n_[j]] ? 1.0f : 0.0f;
#pragma unroll
      for (int i = 0; i < 8; ++i) {
        float v = acc[j][i];
        if (RELU) v = fmaxf(v, 0.0f);
        atomicAdd(&poolAcc[mBase + 8 * lh + i], v * mval);
      }
    }
    __syncthreads();
    if (tid < (unsigned)COUT) {
      const unsigned bb = (blockIdx.x * 64u) / (unsigned)SPT;
      atomicAdd(&pooled[bb * COUT + tid], poolAcc[tid]);
    }
  } else {
#pragma unroll
    for (int j = 0; j < NB; ++j) {
      const float mval = mk[n_[j]] ? 1.0f : 0.0f;
      bf16x8 st;
#pragma unroll
      for (int i = 0; i < 8; ++i) {
        float v = acc[j][i];
        if (RELU) v = fmaxf(v, 0.0f);
        st[i] = (__bf16)(v * mval);
      }
      *(bf16x8*)(out + (size_t)n_[j] * COUT + mBase + 8 * lh) = st;
    }
  }
}

// ---------------------------------------------------------------------------
// head: pooled/cnt -> 128x10 linear -> relu -> softmax   (one block)
// ---------------------------------------------------------------------------
__global__ __launch_bounds__(256) void final_head(
    const float* __restrict__ pooled, const float* __restrict__ cnt,
    const float* __restrict__ wl, const float* __restrict__ bl,
    float* __restrict__ out) {
  __shared__ float lg[16][10];
  int tid = threadIdx.x;
  if (tid < 160) {
    int b = tid / 10, j = tid % 10;
    float s = 0.0f;
#pragma unroll 4
    for (int c = 0; c < 128; ++c) s += pooled[b * 128 + c] * wl[j * 128 + c];
    lg[b][j] = fmaxf(s / cnt[b] + bl[j], 0.0f);
  }
  __syncthreads();
  if (tid < 16) {
    int b = tid;
    float mx = lg[b][0];
#pragma unroll
    for (int j = 1; j < 10; ++j) mx = fmaxf(mx, lg[b][j]);
    float e[10], sum = 0.0f;
#pragma unroll
    for (int j = 0; j < 10; ++j) { e[j] = __expf(lg[b][j] - mx); sum += e[j]; }
#pragma unroll
    for (int j = 0; j < 10; ++j) out[b * 10 + j] = e[j] / sum;
  }
}

// ---------------------------------------------------------------------------
extern "C" void kernel_launch(void* const* d_in, const int* in_sizes, int n_in,
                              void* d_out, int out_size, void* d_ws, size_t ws_size,
                              hipStream_t stream) {
  using namespace nm;
  const float* x  = (const float*)d_in[0];
  const float* w1 = (const float*)d_in[1];
  const float* w2 = (const float*)d_in[2];
  const float* w3 = (const float*)d_in[3];
  const float* wl = (const float*)d_in[4];
  const float* bl = (const float*)d_in[5];
  float* out = (float*)d_out;

  auto al = [](size_t v) { return (v + 255) & ~(size_t)255; };
  char* ws = (char*)d_ws;
  size_t o = 0;
  __bf16* xp  = (__bf16*)(ws + o); o += al((size_t)POS1 * 2 * sizeof(__bf16));
  __bf16* h1  = (__bf16*)(ws + o); o += al((size_t)POS1 * 32 * sizeof(__bf16));
  __bf16* h2  = (__bf16*)(ws + o); o += al((size_t)POS2 * 64 * sizeof(__bf16));
  unsigned char* m1 = (unsigned char*)(ws + o); o += al((size_t)POS1);
  unsigned char* m2 = (unsigned char*)(ws + o); o += al((size_t)POS2);
  float* cnt    = (float*)(ws + o); o += al(16 * sizeof(float));
  float* pooled = (float*)(ws + o); o += al(16 * 128 * sizeof(float));
  __bf16* w1p = (__bf16*)(ws + o); o += al((size_t)32 * 256 * sizeof(__bf16));
  __bf16* w2p = (__bf16*)(ws + o); o += al((size_t)64 * 4000 * sizeof(__bf16));
  __bf16* w3p = (__bf16*)(ws + o); o += al((size_t)128 * 1728 * sizeof(__bf16));

  // prep
  prep_x_mask<<<POS1 / 256, 256, 0, stream>>>(x, xp, m1);
  prep_pool_mask<<<POS2 / 256, 256, 0, stream>>>(m1, m2);
  prep_count_zero<<<16, 256, 0, stream>>>(m2, cnt, pooled);
  prep_w<<<(32 * 256) / 256, 256, 0, stream>>>(w1, w1p, 2, 32, 125, 256);
  prep_w<<<(64 * 4000) / 256, 256, 0, stream>>>(w2, w2p, 32, 64, 125, 4000);
  prep_w<<<(128 * 1728) / 256, 256, 0, stream>>>(w3, w3p, 64, 128, 27, 1728);

  // conv1: 2->32, 5^3, s1 p2, mask, no relu. 2 m-tiles, 4 n-groups/block.
  conv_wmma<2, 32, 5, 5, 5, 1, 2, 128, 34, 34, 128, 34, 34,
            250, 256, 2, false, false>
      <<<POS1 / 256, 256, 0, stream>>>(xp, w1p, m1, h1, nullptr);

  // conv2: 32->64, 5^3, s2 p2, relu, mask2. 4 m-tiles, 2 n-groups/block.
  conv_wmma<32, 64, 5, 5, 5, 2, 2, 128, 34, 34, 64, 17, 17,
            4000, 4000, 4, true, false>
      <<<POS2 / 128, 256, 0, stream>>>(h1, w2p, m2, h2, nullptr);

  // conv3: 64->128, 3^3, s1 p1, relu, mask2, fused global pooling.
  conv_wmma<64, 128, 3, 3, 3, 1, 1, 64, 17, 17, 64, 17, 17,
            1728, 1728, 8, true, true>
      <<<POS2 / 64, 256, 0, stream>>>(h2, w3p, m2, nullptr, pooled);

  // head
  final_head<<<1, 256, 0, stream>>>(pooled, cnt, wl, bl, out);
}
